// Enwik8Model_25443386261683
// MI455X (gfx1250) — matrix operand order repaired and checked by reading, not compiled, not run on hardware
//
#include <hip/hip_runtime.h>
#include <math.h>

// Problem constants (match reference)
#define BB    32
#define SS    1024
#define DIN   256
#define DHH   512
#define DHALF 256
#define DOUT  256
#define NBLK  3
#define NROT  3

// Padded LDS row stride for the 32x512 activation: 528 bf16 = 1056 B = 33*32B.
// Keeps 32-byte alignment for v16bf loads while rotating 8 LDS banks per row
// (plain 1024B stride would put all 16 rows of a half-wave on the same 8 banks).
#define LDSS  528

typedef __bf16 bf16_t;
typedef __attribute__((ext_vector_type(16))) __bf16 v16bf;
typedef __attribute__((ext_vector_type(8)))  float  v8f;

static __device__ __forceinline__ v8f wmma_bf16(v16bf a, v16bf b, v8f c) {
  // D = A(16x32) * B(32x16) + C, f32 accum
  return __builtin_amdgcn_wmma_f32_16x16x32_bf16(false, a, false, b, (short)0, c,
                                                 false, false);
}

static __device__ __forceinline__ void wait_async0() {
#if __has_builtin(__builtin_amdgcn_s_wait_asynccnt)
  __builtin_amdgcn_s_wait_asynccnt(0);
#else
  asm volatile("s_wait_asynccnt 0x0" ::: "memory");
#endif
}

static __device__ __forceinline__ float fast_tanh(float x) {
#if __has_builtin(__builtin_amdgcn_tanhf)
  return __builtin_amdgcn_tanhf(x);
#else
  return tanhf(x);
#endif
}

// ---------------------------------------------------------------------------
// Weight prep: convert f32 weights into bf16 "B-ready" [K][N] layout.
// mode 0: dst[k*N+n] = src[n*K+k]   (W used as u @ W.T)
// mode 1: dst[k*N+n] = src[k*N+n]   (plain, u @ P)
// mode 2: dst[k*N+n] = 0.5*(src[k*N+n] + src[n*N+k])  (symmetrized reflector)
// ---------------------------------------------------------------------------
__global__ __launch_bounds__(256) void pack_w_kernel(const float* __restrict__ src,
                                                     bf16_t* __restrict__ dst,
                                                     int K, int N, int mode) {
  int i = blockIdx.x * 256 + threadIdx.x;
  if (i >= K * N) return;
  int k = i / N, n = i - k * N;
  float v;
  if (mode == 0)      v = src[(size_t)n * K + k];
  else if (mode == 1) v = src[i];
  else                v = 0.5f * (src[i] + src[(size_t)n * N + k]);
  dst[i] = (bf16_t)v;
}

// Elementwise f32 -> bf16 convert (for x)
__global__ __launch_bounds__(256) void convert_bf16_kernel(const float* __restrict__ src,
                                                           bf16_t* __restrict__ dst,
                                                           int n) {
  int i = blockIdx.x * 256 + threadIdx.x;
  if (i < n) dst[i] = (bf16_t)src[i];
}

// ---------------------------------------------------------------------------
// Input projection: h[s][b][n] = sum_k x[b][s][k] * W_in[n][k] + b_in[n]
// xbf bf16 [B][S][DIN], Bin bf16 [DIN][DHH], h bf16 [S][B][DHH]
// Block: 256 threads (8 waves), grid (S, DHH/128). Wave handles one n-tile,
// both m-tiles (m = batch 0..31).
// ---------------------------------------------------------------------------
__global__ __launch_bounds__(256) void proj_in_kernel(const bf16_t* __restrict__ xbf,
                                                      const bf16_t* __restrict__ Bin,
                                                      const float* __restrict__ b_in,
                                                      bf16_t* __restrict__ h) {
  int s    = blockIdx.x;
  int wave = threadIdx.x >> 5;
  int lane = threadIdx.x & 31;
  int n0   = (blockIdx.y * 8 + wave) * 16;

  v8f acc0 = {}, acc1 = {};
  int mlo = lane & 15;
  int khl = (lane >> 4) << 4;

#pragma unroll 4
  for (int kc = 0; kc < DIN; kc += 32) {
    int ks = kc + khl;
    v16bf a0 = *(const v16bf*)(xbf + ((size_t)mlo * SS + s) * DIN + ks);
    v16bf a1 = *(const v16bf*)(xbf + ((size_t)(mlo + 16) * SS + s) * DIN + ks);
    v16bf b;
    for (int e = 0; e < 16; ++e) b[e] = Bin[(size_t)(ks + e) * DHH + n0 + (lane & 15)];
    acc0 = wmma_bf16(a0, b, acc0);
    acc1 = wmma_bf16(a1, b, acc1);
  }
#pragma unroll
  for (int r = 0; r < 8; ++r) {
    int m = r + ((lane >> 4) << 3);
    int n = n0 + (lane & 15);
    float bias = b_in[n];
    h[((size_t)s * BB + m) * DHH + n]      = (bf16_t)(acc0[r] + bias);
    h[((size_t)s * BB + m + 16) * DHH + n] = (bf16_t)(acc1[r] + bias);
  }
}

// ---------------------------------------------------------------------------
// Persistent scan kernel: 1 workgroup, 1024 threads = 32 wave32.
// Activation (32 x 512 bf16, padded stride) ping-pongs between two LDS
// buffers; h_{t+1} is staged to LDS with async global->LDS DMA while step t
// computes. Carry state kept f32 in global scratch (L2-resident).
// ---------------------------------------------------------------------------
static __device__ __forceinline__ void gemm512_acc(const bf16_t* __restrict__ src,
                                                   const bf16_t* __restrict__ W,
                                                   int lane, int wave,
                                                   v8f& acc0, v8f& acc1) {
  int n0 = wave * 16;
  const bf16_t* abase = src + (size_t)(lane & 15) * LDSS + ((lane >> 4) << 4);
#pragma unroll 4
  for (int kc = 0; kc < DHH; kc += 32) {
    // stream next-next B chunk toward WGP$ while WMMAs retire
    if (kc + 64 < DHH)
      __builtin_prefetch(W + (size_t)(kc + 64 + lane) * DHH + n0, 0, 3);
    v16bf a0 = *(const v16bf*)(abase + kc);
    v16bf a1 = *(const v16bf*)(abase + (size_t)16 * LDSS + kc);
    v16bf b;
    for (int e = 0; e < 16; ++e) b[e] = W[(size_t)(kc + ((lane >> 4) << 4) + e) * DHH + n0 + (lane & 15)];
    acc0 = wmma_bf16(a0, b, acc0);
    acc1 = wmma_bf16(a1, b, acc1);
  }
}

static __device__ __forceinline__ void gemm512_stage(const bf16_t* __restrict__ src,
                                                     bf16_t* __restrict__ dst,
                                                     const bf16_t* __restrict__ W,
                                                     int lane, int wave) {
  v8f acc0 = {}, acc1 = {};
  gemm512_acc(src, W, lane, wave, acc0, acc1);
  int n0 = wave * 16;
#pragma unroll
  for (int r = 0; r < 8; ++r) {
    int m = r + ((lane >> 4) << 3);
    int n = n0 + (lane & 15);
    dst[(size_t)m * LDSS + n]        = (bf16_t)acc0[r];
    dst[(size_t)(m + 16) * LDSS + n] = (bf16_t)acc1[r];
  }
  __syncthreads();
}

// One half of a reversible coupling block:
//   dsth[m][n] = resid[m][n] + tanh( (srcg @ W)[m][n] + bias[n] )
// srcg/resid/dsth are column-offset views into 32x512 buffers (row stride LDSS).
static __device__ __forceinline__ void rev_half(const bf16_t* __restrict__ srcg,
                                                const bf16_t* __restrict__ resid,
                                                bf16_t* __restrict__ dsth,
                                                const bf16_t* __restrict__ W,
                                                const float* __restrict__ bias,
                                                int lane, int wave) {
  int nt = wave & 15, mt = wave >> 4;
  int n0 = nt * 16, m0 = mt * 16;
  v8f acc = {};
  const bf16_t* abase = srcg + (size_t)(m0 + (lane & 15)) * LDSS + ((lane >> 4) << 4);
#pragma unroll 4
  for (int kc = 0; kc < DHALF; kc += 32) {
    if (kc + 64 < DHALF)
      __builtin_prefetch(W + (size_t)(kc + 64 + lane) * DHALF + n0, 0, 3);
    v16bf a = *(const v16bf*)(abase + kc);
    v16bf b;
    for (int e = 0; e < 16; ++e) b[e] = W[(size_t)(kc + ((lane >> 4) << 4) + e) * DHALF + n0 + (lane & 15)];
    acc = wmma_bf16(a, b, acc);
  }
#pragma unroll
  for (int r = 0; r < 8; ++r) {
    int m = m0 + r + ((lane >> 4) << 3);
    int n = n0 + (lane & 15);
    float v = fast_tanh(acc[r] + bias[n]) + (float)resid[(size_t)m * LDSS + n];
    dsth[(size_t)m * LDSS + n] = (bf16_t)v;
  }
  __syncthreads();
}

__global__ __launch_bounds__(1024) void enigma_scan_kernel(
    const bf16_t* __restrict__ h,      // [S][B][DHH]
    const bf16_t* __restrict__ Bp1,    // P^T            [DHH][DHH]
    const bf16_t* __restrict__ Brot,   // rotors^T   [3][DHH][DHH]
    const bf16_t* __restrict__ Bwf,    // Wf^T       [3][DHALF][DHALF]
    const bf16_t* __restrict__ Bwg,    // Wg^T       [3][DHALF][DHALF]
    const float*  __restrict__ bfv,    // [3][DHALF]
    const float*  __restrict__ bgv,    // [3][DHALF]
    const bf16_t* __restrict__ Brefl,  // 0.5(A+A^T)     [DHH][DHH]
    const bf16_t* __restrict__ Bp2,    // P (plain)      [DHH][DHH]
    float*        __restrict__ stateg, // [B][DHH] f32 carry
    bf16_t*       __restrict__ uh)     // [S][B][DHH] history for out-proj
{
  extern __shared__ bf16_t smem[];
  bf16_t* buf0 = smem;                     // 32 x LDSS (padded)
  bf16_t* buf1 = smem + BB * LDSS;         // 32 x LDSS (padded)
  bf16_t* hs   = smem + 2 * BB * LDSS;     // h staging, plain [B*DHH]
  const unsigned HS_BYTE_OFF = (unsigned)(2 * BB * LDSS * sizeof(bf16_t));

  int tid  = threadIdx.x;
  int lane = tid & 31;
  int wave = tid >> 5;

  for (int t = 0; t < SS; ++t) {
    bf16_t* cur = buf0;
    bf16_t* nxt = buf1;

    // ---- make staged h_t visible to every wave ----
    wait_async0();
    __syncthreads();

    // ---- pack: v = h_t + state -> cur (bf16, padded stride) ----
    const bf16_t* hrow = h + (size_t)t * BB * DHH;
    for (int i = tid; i < BB * DHH; i += 1024) {
      float hv = (t == 0) ? (float)hrow[i] : (float)hs[i];
      float st = (t == 0) ? 0.0f : stateg[i];
      int m = i >> 9, n = i & (DHH - 1);
      cur[m * LDSS + n] = (bf16_t)(hv + st);
    }
    __syncthreads();

    // ---- kick off async DMA of h_{t+1} into LDS (overlaps 18 GEMM stages) --
    if (t + 1 < SS) {
      unsigned long long gbase =
          (unsigned long long)(const void*)(h + (size_t)(t + 1) * BB * DHH);
      for (int c = tid; c < (BB * DHH) / 8; c += 1024) {  // 16B chunks
        unsigned goff = (unsigned)(c * 16);
        unsigned loff = HS_BYTE_OFF + (unsigned)(c * 16);
        asm volatile("global_load_async_to_lds_b128 %0, %1, %2"
                     :: "v"(loff), "v"(goff), "s"(gbase)
                     : "memory");
      }
    }

    // ---- plugboard: u = v @ P.T ----
    gemm512_stage(cur, nxt, Bp1, lane, wave);
    { bf16_t* tmp = cur; cur = nxt; nxt = tmp; }

    // ---- rotors ----
    for (int i = 0; i < NROT; ++i) {
      gemm512_stage(cur, nxt, Brot + (size_t)i * DHH * DHH, lane, wave);
      bf16_t* tmp = cur; cur = nxt; nxt = tmp;
    }

    // ---- reversible blocks (forward) ----
    for (int i = 0; i < NBLK; ++i) {
      const bf16_t* Wf = Bwf + (size_t)i * DHALF * DHALF;
      const bf16_t* Wg = Bwg + (size_t)i * DHALF * DHALF;
      rev_half(cur + DHALF, cur,         nxt,         Wf, bfv + i * DHALF, lane, wave);
      rev_half(nxt,         cur + DHALF, nxt + DHALF, Wg, bgv + i * DHALF, lane, wave);
      bf16_t* tmp = cur; cur = nxt; nxt = tmp;
    }

    // ---- reflector ----
    gemm512_stage(cur, nxt, Brefl, lane, wave);
    { bf16_t* tmp = cur; cur = nxt; nxt = tmp; }

    // ---- reversible blocks (reverse order) ----
    for (int i = NBLK - 1; i >= 0; --i) {
      const bf16_t* Wf = Bwf + (size_t)i * DHALF * DHALF;
      const bf16_t* Wg = Bwg + (size_t)i * DHALF * DHALF;
      rev_half(cur + DHALF, cur,         nxt,         Wf, bfv + i * DHALF, lane, wave);
      rev_half(nxt,         cur + DHALF, nxt + DHALF, Wg, bgv + i * DHALF, lane, wave);
      bf16_t* tmp = cur; cur = nxt; nxt = tmp;
    }

    // ---- plugboard transpose: u = u @ P ; write carry (f32) + history (bf16)
    {
      v8f acc0 = {}, acc1 = {};
      gemm512_acc(cur, Bp2, lane, wave, acc0, acc1);
      int n0 = wave * 16;
      bf16_t* urow = uh + (size_t)t * BB * DHH;
#pragma unroll
      for (int r = 0; r < 8; ++r) {
        int m = r + ((lane >> 4) << 3);
        int n = n0 + (lane & 15);
        stateg[(size_t)m * DHH + n]        = acc0[r];
        stateg[(size_t)(m + 16) * DHH + n] = acc1[r];
        urow[(size_t)m * DHH + n]          = (bf16_t)acc0[r];
        urow[(size_t)(m + 16) * DHH + n]   = (bf16_t)acc1[r];
      }
      __threadfence();
      __syncthreads();
    }
  }
}

// ---------------------------------------------------------------------------
// Output projection: out[b][s][n] = sum_k uh[s][b][k] * W_out[n][k] + b_out[n]
// uh bf16 [S][B][DHH], Bout bf16 [DHH][DOUT], out f32 [B][S][DOUT]
// ---------------------------------------------------------------------------
__global__ __launch_bounds__(256) void proj_out_kernel(const bf16_t* __restrict__ uh,
                                                       const bf16_t* __restrict__ Bout,
                                                       const float* __restrict__ b_out,
                                                       float* __restrict__ out) {
  int s    = blockIdx.x;
  int wave = threadIdx.x >> 5;
  int lane = threadIdx.x & 31;
  int n0   = (blockIdx.y * 8 + wave) * 16;

  v8f acc0 = {}, acc1 = {};
  const bf16_t* abase = uh + ((size_t)s * BB + (lane & 15)) * DHH + ((lane >> 4) << 4);
#pragma unroll 4
  for (int kc = 0; kc < DHH; kc += 32) {
    v16bf a0 = *(const v16bf*)(abase + kc);
    v16bf a1 = *(const v16bf*)(abase + (size_t)16 * DHH + kc);
    v16bf b;
    for (int e = 0; e < 16; ++e) b[e] = Bout[(size_t)(kc + ((lane >> 4) << 4) + e) * DOUT + n0 + (lane & 15)];
    acc0 = wmma_bf16(a0, b, acc0);
    acc1 = wmma_bf16(a1, b, acc1);
  }
#pragma unroll
  for (int r = 0; r < 8; ++r) {
    int m = r + ((lane >> 4) << 3);
    int n = n0 + (lane & 15);
    out[((size_t)m * SS + s) * DOUT + n]        = acc0[r] + b_out[n];
    out[((size_t)(m + 16) * SS + s) * DOUT + n] = acc1[r] + b_out[n];
  }
}

// ---------------------------------------------------------------------------
// Host orchestration
// ---------------------------------------------------------------------------
extern "C" void kernel_launch(void* const* d_in, const int* in_sizes, int n_in,
                              void* d_out, int out_size, void* d_ws, size_t ws_size,
                              hipStream_t stream) {
  (void)in_sizes; (void)n_in; (void)out_size; (void)ws_size;

  const float* x      = (const float*)d_in[0];   // [B][S][DIN]
  const float* W_in   = (const float*)d_in[1];   // [DHH][DIN]
  const float* b_in   = (const float*)d_in[2];   // [DHH]
  const float* P      = (const float*)d_in[3];   // [DHH][DHH]
  const float* rotors = (const float*)d_in[4];   // [3][DHH][DHH]
  const float* Wf     = (const float*)d_in[5];   // [3][DHALF][DHALF]
  const float* bfv    = (const float*)d_in[6];   // [3][DHALF]
  const float* Wg     = (const float*)d_in[7];   // [3][DHALF][DHALF]
  const float* bgv    = (const float*)d_in[8];   // [3][DHALF]
  const float* A      = (const float*)d_in[9];   // [DHH][DHH]
  const float* W_out  = (const float*)d_in[10];  // [DOUT][DHH]
  const float* b_out  = (const float*)d_in[11];  // [DOUT]
  float* out = (float*)d_out;

  // Workspace layout (bytes), all chunks naturally aligned
  char* ws = (char*)d_ws;
  size_t off = 0;
  auto alloc = [&](size_t bytes) { char* p = ws + off; off += (bytes + 255) & ~(size_t)255; return p; };

  bf16_t* Bin    = (bf16_t*)alloc((size_t)DIN * DHH * 2);
  bf16_t* Bp1    = (bf16_t*)alloc((size_t)DHH * DHH * 2);
  bf16_t* Brot   = (bf16_t*)alloc((size_t)NROT * DHH * DHH * 2);
  bf16_t* Bwf    = (bf16_t*)alloc((size_t)NBLK * DHALF * DHALF * 2);
  bf16_t* Bwg    = (bf16_t*)alloc((size_t)NBLK * DHALF * DHALF * 2);
  bf16_t* Brefl  = (bf16_t*)alloc((size_t)DHH * DHH * 2);
  bf16_t* Bp2    = (bf16_t*)alloc((size_t)DHH * DHH * 2);
  bf16_t* Bout   = (bf16_t*)alloc((size_t)DHH * DOUT * 2);
  bf16_t* xbf    = (bf16_t*)alloc((size_t)BB * SS * DIN * 2);
  bf16_t* hbuf   = (bf16_t*)alloc((size_t)SS * BB * DHH * 2);
  bf16_t* uh     = (bf16_t*)alloc((size_t)SS * BB * DHH * 2);
  float*  stateg = (float*) alloc((size_t)BB * DHH * 4);

  auto blocks = [](int n) { return (n + 255) / 256; };

  // --- weight prep (parallel, tiny) ---
  pack_w_kernel<<<blocks(DIN * DHH),  256, 0, stream>>>(W_in,  Bin,  DIN, DHH, 0);
  pack_w_kernel<<<blocks(DHH * DHH),  256, 0, stream>>>(P,     Bp1,  DHH, DHH, 0);
  for (int i = 0; i < NROT; ++i)
    pack_w_kernel<<<blocks(DHH * DHH), 256, 0, stream>>>(rotors + (size_t)i * DHH * DHH,
                                                         Brot + (size_t)i * DHH * DHH, DHH, DHH, 0);
  for (int i = 0; i < NBLK; ++i) {
    pack_w_kernel<<<blocks(DHALF * DHALF), 256, 0, stream>>>(Wf + (size_t)i * DHALF * DHALF,
                                                             Bwf + (size_t)i * DHALF * DHALF,
                                                             DHALF, DHALF, 0);
    pack_w_kernel<<<blocks(DHALF * DHALF), 256, 0, stream>>>(Wg + (size_t)i * DHALF * DHALF,
                                                             Bwg + (size_t)i * DHALF * DHALF,
                                                             DHALF, DHALF, 0);
  }
  pack_w_kernel<<<blocks(DHH * DHH),  256, 0, stream>>>(A,     Brefl, DHH, DHH, 2);
  pack_w_kernel<<<blocks(DHH * DHH),  256, 0, stream>>>(P,     Bp2,   DHH, DHH, 1);
  pack_w_kernel<<<blocks(DHH * DOUT), 256, 0, stream>>>(W_out, Bout,  DHH, DOUT, 0);
  convert_bf16_kernel<<<blocks(BB * SS * DIN), 256, 0, stream>>>(x, xbf, BB * SS * DIN);

  // --- input projection (parallel WMMA GEMM): h = x @ W_in.T + b_in ---
  proj_in_kernel<<<dim3(SS, DHH / 128), 256, 0, stream>>>(xbf, Bin, b_in, hbuf);

  // --- serial Enigma scan (persistent single workgroup, 32 wave32) ---
  const int smem_bytes = (2 * BB * LDSS + BB * DHH) * (int)sizeof(bf16_t);
  (void)hipFuncSetAttribute(reinterpret_cast<const void*>(enigma_scan_kernel),
                            hipFuncAttributeMaxDynamicSharedMemorySize, smem_bytes);
  enigma_scan_kernel<<<1, 1024, smem_bytes, stream>>>(hbuf, Bp1, Brot, Bwf, Bwg,
                                                      bfv, bgv, Brefl, Bp2,
                                                      stateg, uh);

  // --- output projection (parallel WMMA GEMM): out = u @ W_out.T + b_out ---
  proj_out_kernel<<<dim3(SS, DOUT / 128), 256, 0, stream>>>(uh, Bout, b_out, out);
}
